// SoftRAMAttentionV2_88510686036789
// MI455X (gfx1250) — matrix operand order repaired and checked by reading, not compiled, not run on hardware
//
#include <hip/hip_runtime.h>

// Problem constants (match reference)
#define S   128
#define B   256
#define H   16
#define NB  12
#define LUT 4096        // 2^NB
#define WORDS 128       // LUT/32 sign-bitmap words per (h,n)
#define HN  (H * B)     // 4096 (h,n) pairs
#define NT  16          // neurons per block in main kernel

typedef __attribute__((ext_vector_type(16))) _Float16 v16h;
typedef __attribute__((ext_vector_type(8)))  float    v8f;
typedef __attribute__((ext_vector_type(4)))  int      v4i;

// gfx1250 async global->LDS copy path (ASYNCcnt-tracked, bypasses VGPRs).
// Probe round 2 revealed the builtin's param types: (global int4*, lds int4*, imm, imm).
#if __has_builtin(__builtin_amdgcn_global_load_async_to_lds_b128) && \
    __has_builtin(__builtin_amdgcn_s_wait_asynccnt)
#define HAVE_ASYNC_LDS 1
__device__ __forceinline__ void async_cp16(void* ldsdst, const void* gsrc) {
    __builtin_amdgcn_global_load_async_to_lds_b128(
        (__attribute__((address_space(1))) v4i*)(gsrc),
        (__attribute__((address_space(3))) v4i*)(ldsdst),
        /*offset=*/0, /*cpol=*/0);
}
#else
#define HAVE_ASYNC_LDS 0
#endif

// ---------------------------------------------------------------------------
// Kernel 1: pack sign bits of memory[h][n][addr] into bitmap words.
// One thread per output word; 8x float4 loads (the single 64MB HBM pass).
// ---------------------------------------------------------------------------
__global__ void k_signs(const float* __restrict__ mem, unsigned* __restrict__ signw) {
    int w = blockIdx.x * blockDim.x + threadIdx.x;       // word index over HN*WORDS
    if (w >= HN * WORDS) return;
    const float4* p = (const float4*)(mem + (size_t)w * 32);
    unsigned m = 0;
#pragma unroll
    for (int q = 0; q < 8; ++q) {
        float4 v = p[q];
        m |= (v.x > 0.0f ? 1u : 0u) << (4 * q + 0);
        m |= (v.y > 0.0f ? 1u : 0u) << (4 * q + 1);
        m |= (v.z > 0.0f ? 1u : 0u) << (4 * q + 2);
        m |= (v.w > 0.0f ? 1u : 0u) << (4 * q + 3);
    }
    signw[w] = m;
}

// ---------------------------------------------------------------------------
// Kernel 2: separable address components. connections bits split by range:
//   c <  256        -> query bit  (depends on i)
//   256 <= c < 512  -> key bit    (depends on j)
//   c >= 512        -> rel-pos bit (depends on r = i-j, 0..127)
// addr(i,j) = Aq[hn][i] | Ak[hn][j] | Ap[hn][i-j]
// ---------------------------------------------------------------------------
__global__ void k_addr(const int* __restrict__ tokens, const int* __restrict__ conn,
                       unsigned short* __restrict__ Aq, unsigned short* __restrict__ Ak,
                       unsigned short* __restrict__ Ap) {
    int gid = blockIdx.x * blockDim.x + threadIdx.x;     // over HN*S
    if (gid >= HN * S) return;
    int x  = gid & (S - 1);                              // plays role of i, j, or r
    int hn = gid >> 7;
    const int* c = conn + hn * NB;
    unsigned aq = 0, ak = 0, ap = 0;
#pragma unroll
    for (int b = 0; b < NB; ++b) {
        int cc = c[b];
        if (cc < B) {
            aq |= ((unsigned)tokens[x * B + cc] & 1u) << b;
        } else if (cc < 2 * B) {
            ak |= ((unsigned)tokens[x * B + (cc - B)] & 1u) << b;
        } else {
            ap |= ((unsigned)((x >> (cc - 2 * B)) & 1)) << b;
        }
    }
    Aq[gid] = (unsigned short)aq;
    Ak[gid] = (unsigned short)ak;
    Ap[gid] = (unsigned short)ap;
}

// ---------------------------------------------------------------------------
// Kernel 3: main gather/XOR loop. Block = (head h, tile of NT neurons).
// Per-block working set is contiguous in global AND in LDS, so staging is a
// straight copy: gfx1250 GLOBAL_LOAD_ASYNC_TO_LDS_B128 + S_WAIT_ASYNCCNT.
// Wave-uniform neuron row -> broadcast LDS reads for Ak; consecutive-lane
// reads for Ap. Writes head bits as f16 into hr[(i*B+n)*H + h].
// ---------------------------------------------------------------------------
__global__ void k_main(const unsigned* __restrict__ signw,
                       const unsigned short* __restrict__ Aq,
                       const unsigned short* __restrict__ Ak,
                       const unsigned short* __restrict__ Ap,
                       _Float16* __restrict__ hr) {
    __shared__ unsigned       lds_sign[NT][WORDS];  // 8 KB, contiguous
    __shared__ unsigned short lds_aq[NT][S];        // 4 KB, contiguous
    __shared__ unsigned short lds_ak[NT][S];        // 4 KB, contiguous
    __shared__ unsigned short lds_ap[NT][S];        // 4 KB, contiguous

    int tid   = threadIdx.x;                        // 0..255
    int h     = blockIdx.x / (B / NT);              // 0..15
    int ntile = blockIdx.x % (B / NT);              // 0..15
    int n0    = ntile * NT;
    int hnb   = h * B + n0;

#if HAVE_ASYNC_LDS
    {
        // Sign bitmap tile: 8 KB contiguous -> 512 b128 async copies.
        const char* gs = (const char*)(signw + (size_t)hnb * WORDS);
        char*       ls = (char*)&lds_sign[0][0];
        for (int f = tid; f < (NT * WORDS * 4) / 16; f += 256)
            async_cp16(ls + f * 16, gs + f * 16);
        // Address tables: 4 KB each -> 256 b128 async copies apiece.
        const char* gq = (const char*)(Aq + (size_t)hnb * S);
        const char* gk = (const char*)(Ak + (size_t)hnb * S);
        const char* gp = (const char*)(Ap + (size_t)hnb * S);
        char* lq = (char*)&lds_aq[0][0];
        char* lk = (char*)&lds_ak[0][0];
        char* lp = (char*)&lds_ap[0][0];
        for (int f = tid; f < (NT * S * 2) / 16; f += 256) {
            async_cp16(lq + f * 16, gq + f * 16);
            async_cp16(lk + f * 16, gk + f * 16);
            async_cp16(lp + f * 16, gp + f * 16);
        }
        __builtin_amdgcn_s_wait_asynccnt(0);
    }
#else
    for (int f = tid; f < NT * WORDS; f += 256) {
        int t = f >> 7, w = f & (WORDS - 1);
        lds_sign[t][w] = signw[(size_t)(hnb + t) * WORDS + w];
    }
    for (int f = tid; f < NT * S; f += 256) {
        int t = f >> 7, x = f & (S - 1);
        int src = (hnb + t) * S + x;
        lds_aq[t][x] = Aq[src];
        lds_ak[t][x] = Ak[src];
        lds_ap[t][x] = Ap[src];
    }
#endif
    __syncthreads();

    // 2048 (i,n) pairs per block, 8 per thread. Within a wave: n uniform,
    // i consecutive. Alternate i direction per step for load balance.
#pragma unroll 1
    for (int s = 0; s < 8; ++s) {
        int i = tid & (S - 1);
        if (s & 1) i = (S - 1) - i;
        int n = (tid >> 7) + 2 * s;                 // 0..15, wave-uniform
        unsigned aq  = lds_aq[n][i];
        unsigned acc = 0;
        for (int j = 0; j <= i; ++j) {
            unsigned addr = aq | (unsigned)lds_ak[n][j] | (unsigned)lds_ap[n][i - j];
            unsigned w    = lds_sign[n][addr >> 5];
            acc ^= (w >> (addr & 31));
        }
        acc &= 1u;
        hr[(size_t)(i * B + n0 + n) * H + h] = (_Float16)(float)acc;
    }
}

// ---------------------------------------------------------------------------
// Kernel 4: majority vote over heads via v_wmma_f32_16x16x32_f16.
// Heads are the K dimension. A = ones for all K (all 16 halves = 1.0 in
// EVERY lane). B: lane L<16 carries column L with
// K=0..15 = head bits (one contiguous 32B v16h load); lanes 16..31 carry
// K=16..31 = zeros. All D rows equal sum over heads; lane<16 thresholds >8.
// ---------------------------------------------------------------------------
__global__ void k_reduce(const _Float16* __restrict__ hr, float* __restrict__ out) {
    int tid  = threadIdx.x;
    int lane = tid & 31;
    int wave = (blockIdx.x * blockDim.x + tid) >> 5;   // global wave id
    int base = wave * 16;                              // 16 outputs per wave

    v16h a;
#pragma unroll
    for (int e = 0; e < 16; ++e) a[e] = (_Float16)1.0f;

    int col = lane & 15;
    const v16h* bp = (const v16h*)(hr + (size_t)(base + col) * H);
    v16h bv = *bp;                                     // valid address for all lanes
    v16h bz = {};
    bv = (lane < 16) ? bv : bz;                        // K=16..31 -> zero, no divergence

    v8f c = {};
    c = __builtin_amdgcn_wmma_f32_16x16x32_f16(
            /*neg_a=*/false, a, /*neg_b=*/false, bv,
            /*c_mod=*/(short)0, c, /*reuse_a=*/false, /*reuse_b=*/false);

    if (lane < 16) out[base + lane] = (c[0] > 8.5f) ? 1.0f : 0.0f;
}

// ---------------------------------------------------------------------------
extern "C" void kernel_launch(void* const* d_in, const int* in_sizes, int n_in,
                              void* d_out, int out_size, void* d_ws, size_t ws_size,
                              hipStream_t stream) {
    const int*   tokens = (const int*)d_in[0];    // [S,B] int32
    const float* memory = (const float*)d_in[1];  // [H,B,4096] float32
    const int*   conn   = (const int*)d_in[2];    // [H,B,NB] int32
    float*       out    = (float*)d_out;          // [S,B] as float

    char* ws = (char*)d_ws;
    unsigned*       signw = (unsigned*)(ws);                       // 2 MB
    unsigned short* Aq    = (unsigned short*)(ws + (2u << 20));    // 1 MB
    unsigned short* Ak    = (unsigned short*)(ws + (3u << 20));    // 1 MB
    unsigned short* Ap    = (unsigned short*)(ws + (4u << 20));    // 1 MB
    _Float16*       hr    = (_Float16*)(ws + (5u << 20));          // 1 MB

    (void)in_sizes; (void)n_in; (void)out_size; (void)ws_size;

    // 1) 64MB -> 2MB sign bitmap (the only HBM-bound pass)
    k_signs<<<(HN * WORDS + 255) / 256, 256, 0, stream>>>(memory, signw);

    // 2) separable address tables
    k_addr<<<(HN * S + 255) / 256, 256, 0, stream>>>(tokens, conn, Aq, Ak, Ap);

    // 3) LDS-resident gather/XOR core: 256 blocks = 16 heads x 16 neuron tiles
    k_main<<<H * (B / NT), 256, 0, stream>>>(signw, Aq, Ak, Ap, hr);

    // 4) WMMA head-majority reduction: 2048 waves x 16 outputs
    k_reduce<<<(S * B / 16) / 8, 256, 0, stream>>>(hr, out);
}
